// Decoder_5875515261520
// MI455X (gfx1250) — matrix operand order repaired and checked by reading, not compiled, not run on hardware
//
#include <hip/hip_runtime.h>

typedef __attribute__((ext_vector_type(16))) _Float16 v16h;
typedef __attribute__((ext_vector_type(8)))  _Float16 v8h;
typedef __attribute__((ext_vector_type(4)))  _Float16 v4h;
typedef __attribute__((ext_vector_type(8)))  float    v8f;
typedef __attribute__((ext_vector_type(4)))  float    v4f;

#define NHID 64
#define KDIM 128   // 2 * NHID (concat of both endpoints)

// ---- one-time z (f32) -> zh (f16) conversion into workspace ----
__global__ __launch_bounds__(256) void cvt_z_f16_kernel(
    const float* __restrict__ z, _Float16* __restrict__ zh, int n4)
{
    const int i = blockIdx.x * 256 + threadIdx.x;
    if (i < n4) {
        const v4f f = ((const v4f*)z)[i];
        v4h h;
        #pragma unroll
        for (int j = 0; j < 4; ++j) h[j] = (_Float16)f[j];
        ((v4h*)zh)[i] = h;
    }
}

// One wave computes a 16-edge tile:
//   h[16 x 128] (gathered f16)  x  W1^T[128 x 64] (f16 B-fragments, LDS->regs)
//   -> h1[16 x 64] f32 accumulators via 16x v_wmma_f32_16x16x32_f16
//   -> bias + relu + dot(W2) + b2 epilogue (VALU co-executes with WMMA pipe).
template<bool F16Z>
__global__ __launch_bounds__(256) void decoder_edge_mlp_kernel(
    const float*    __restrict__ z,   // [N_NODES, 64] (used when !F16Z)
    const _Float16* __restrict__ zh,  // [N_NODES, 64] (used when F16Z)
    const int*   __restrict__ row,    // [E]
    const int*   __restrict__ col,    // [E]
    const float* __restrict__ W1,     // [64, 128] row-major (out, in)
    const float* __restrict__ b1,     // [64]
    const float* __restrict__ W2,     // [1, 64]
    const float* __restrict__ b2,     // [1]
    float* __restrict__ out,          // [E]
    int E)
{
    // W1 in WMMA B-fragment layout: frag f = nt*4 + kt; lane l; elem j (0..15)
    //   N = nt*16 + (l&15);  K = kt*32 + (l>>4)*8 + (j&7) + (j>>3)*16;  value = W1[N*128 + K]
    __shared__ __align__(32) _Float16 ldsB[16 * 32 * 16]; // 16 KB
    __shared__ float ldsB1[NHID];
    __shared__ float ldsW2[NHID];

    const int t = threadIdx.x;

    // ---- cooperative stage: convert W1 (f32) -> f16 B-fragments in LDS ----
    for (int i = t; i < 16 * 32 * 16; i += 256) {
        const int f  = i >> 9;          // fragment index (512 halfs each)
        const int l  = (i >> 4) & 31;   // lane
        const int j  = i & 15;          // element within lane
        const int nt = f >> 2;
        const int kt = f & 3;
        const int n  = nt * 16 + (l & 15);
        const int k  = kt * 32 + ((l >> 4) << 3) + (j & 7) + ((j >> 3) << 4);
        ldsB[i] = (_Float16)W1[n * KDIM + k];
    }
    if (t < NHID) { ldsB1[t] = b1[t]; ldsW2[t] = W2[t]; }
    __syncthreads();

    const int lane = t & 31;
    const int wave = t >> 5;
    const int m    = lane & 15;   // A-matrix row (edge within tile) / C column N
    const int hi   = lane >> 4;   // half-wave select
    const int kb   = hi * 8;      // A-fragment K base within a 32-wide K tile
    const float bias2 = b2[0];

    const int numTiles = (E + 15) >> 4;
    const int gstride  = gridDim.x * 8;

    for (int tile = blockIdx.x * 8 + wave; tile < numTiles; tile += gstride) {
        const int ebase = tile << 4;
        int eIdx = ebase + m;
        if (eIdx >= E) eIdx = E - 1;           // clamp; store is guarded below
        const long r = row[eIdx];
        const long c = col[eIdx];

        v8f acc[4];
        {
            v8f zero = {};
            acc[0] = zero; acc[1] = zero; acc[2] = zero; acc[3] = zero;
        }

        #pragma unroll
        for (int kt = 0; kt < 4; ++kt) {
            // Global K range of this tile: [kt*32, kt*32+32). K<64 -> z[row], else z[col].
            const int o0 = ((kt & 1) << 5) + kb;   // two 8-elem runs: o0.. , o0+16..
            v16h a;
            if constexpr (F16Z) {
                const _Float16* src = zh + ((kt < 2) ? r : c) * NHID;
                union { v16h v; v8h h[2]; } u;
                u.h[0] = *(const v8h*)(src + o0);        // 16B packed halfs
                u.h[1] = *(const v8h*)(src + o0 + 16);   // 16B packed halfs
                a = u.v;
            } else {
                const float* src = z + ((kt < 2) ? r : c) * NHID;
                const v4f f0 = *(const v4f*)(src + o0);
                const v4f f1 = *(const v4f*)(src + o0 + 4);
                const v4f f2 = *(const v4f*)(src + o0 + 16);
                const v4f f3 = *(const v4f*)(src + o0 + 20);
                #pragma unroll
                for (int j = 0; j < 4; ++j) {
                    a[j]      = (_Float16)f0[j];
                    a[4 + j]  = (_Float16)f1[j];
                    a[8 + j]  = (_Float16)f2[j];
                    a[12 + j] = (_Float16)f3[j];
                }
            }
            #pragma unroll
            for (int nt = 0; nt < 4; ++nt) {
                const v16h b = *(const v16h*)&ldsB[(((nt << 2) | kt) << 9) + (lane << 4)];
                acc[nt] = __builtin_amdgcn_wmma_f32_16x16x32_f16(
                    /*neg_a=*/false, a, /*neg_b=*/false, b,
                    /*c_mod=*/(short)0, acc[nt],
                    /*reuse_a=*/false, /*reuse_b=*/false);
            }
        }

        // ---- layer 2: out[e] = sum_n relu(h1[e][n] + b1[n]) * W2[n] + b2 ----
        // C layout: acc[nt][v] @ lane l == h1[M = v + 8*hi][N = nt*16 + (l&15)]
        float partial[8];
        #pragma unroll
        for (int v = 0; v < 8; ++v) partial[v] = 0.f;
        #pragma unroll
        for (int nt = 0; nt < 4; ++nt) {
            const int n  = nt * 16 + m;
            const float bb = ldsB1[n];
            const float ww = ldsW2[n];
            #pragma unroll
            for (int v = 0; v < 8; ++v) {
                float h = fmaxf(acc[nt][v] + bb, 0.f);
                partial[v] = fmaf(h, ww, partial[v]);
            }
        }
        // reduce across the 16 N-lanes (xor masks < 16 stay within each half-wave)
        #pragma unroll
        for (int v = 0; v < 8; ++v) {
            float s = partial[v];
            s += __shfl_xor(s, 1, 32);
            s += __shfl_xor(s, 2, 32);
            s += __shfl_xor(s, 4, 32);
            s += __shfl_xor(s, 8, 32);
            partial[v] = s;
        }
        if (m == 0) {  // lanes 0 (edges 0..7) and 16 (edges 8..15)
            #pragma unroll
            for (int v = 0; v < 8; ++v) {
                const int e = ebase + hi * 8 + v;
                if (e < E) out[e] = partial[v] + bias2;
            }
        }
    }
}

extern "C" void kernel_launch(void* const* d_in, const int* in_sizes, int n_in,
                              void* d_out, int out_size, void* d_ws, size_t ws_size,
                              hipStream_t stream) {
    const float* z   = (const float*)d_in[0];
    const int*   row = (const int*)d_in[1];
    const int*   col = (const int*)d_in[2];
    const float* W1  = (const float*)d_in[3];
    const float* b1  = (const float*)d_in[4];
    const float* W2  = (const float*)d_in[5];
    const float* b2  = (const float*)d_in[6];
    float* out = (float*)d_out;

    const int E  = in_sizes[1];                 // edge count
    const int nz = in_sizes[0];                 // node-feature element count
    const int numTiles = (E + 15) / 16;
    int blocks = (numTiles + 7) / 8;            // 8 waves/block, 1 tile/wave/iter
    if (blocks > 3072) blocks = 3072;           // amortize W1 staging via grid-stride
    if (blocks < 1) blocks = 1;

    const size_t zhBytes = (size_t)nz * sizeof(_Float16);
    if (ws_size >= zhBytes && (nz & 3) == 0) {
        _Float16* zh = (_Float16*)d_ws;
        const int n4 = nz >> 2;
        cvt_z_f16_kernel<<<(n4 + 255) / 256, 256, 0, stream>>>(z, zh, n4);
        decoder_edge_mlp_kernel<true><<<blocks, 256, 0, stream>>>(
            z, zh, row, col, W1, b1, W2, b2, out, E);
    } else {
        decoder_edge_mlp_kernel<false><<<blocks, 256, 0, stream>>>(
            z, nullptr, row, col, W1, b1, W2, b2, out, E);
    }
}
